// EdgeTypeTransformerConv_66675072303326
// MI455X (gfx1250) — matrix operand order repaired and checked by reading, not compiled, not run on hardware
//
#include <hip/hip_runtime.h>
#include <hip/hip_bf16.h>
#include <math.h>

#define NN 50000
#define EE 800000
#define HC 256

typedef __attribute__((ext_vector_type(16))) __bf16        v16bf;
typedef __attribute__((ext_vector_type(8)))  float          v8f;
typedef __attribute__((ext_vector_type(16))) unsigned short u16x16;
typedef __attribute__((ext_vector_type(8)))  unsigned short u16x8;

// ---- workspace layout (bytes) ----
static constexpr size_t OFF_CNT   = 0;                       // N*4 u32      =    800,000
static constexpr size_t OFF_XB    = 800000;                  // N*256 bf16   = 25,600,000
static constexpr size_t OFF_WT    = 26400000;                // 4*256*256 bf16 = 524,288
static constexpr size_t OFF_Q     = 26924288;                // N*256 f32    = 51,200,000
static constexpr size_t OFF_K     = 78124288;
static constexpr size_t OFF_V     = 129324288;
static constexpr size_t OFF_ALPHA = 180524288;               // E*4 f32      = 12,800,000
static constexpr size_t OFF_AMAX  = 193324288;               // N*4 f32
static constexpr size_t OFF_DEN   = 194124288;               // N*4 f32
static constexpr size_t OFF_WV    = 194924288;               // N*256 f32

__device__ __forceinline__ unsigned short f2bf(float f) {
  union { float f; unsigned u; } v; v.f = f;
  unsigned r = v.u + 0x7FFFu + ((v.u >> 16) & 1u);   // RNE
  return (unsigned short)(r >> 16);
}

__device__ __forceinline__ void atomicMaxF32(float* addr, float val) {
  if (val >= 0.0f) atomicMax((int*)addr, __float_as_int(val));
  else             atomicMin((unsigned int*)addr, __float_as_uint(val));
}

// ---- 1. zero/init accumulators ----
__global__ void init_kernel(unsigned* cnt, float* denom, float* amax, float* wv) {
  int idx = blockIdx.x * 256 + threadIdx.x;
  if (idx < NN * HC) wv[idx] = 0.0f;
  if (idx < NN * 4) { cnt[idx] = 0u; denom[idx] = 0.0f; amax[idx] = -__builtin_inff(); }
}

// ---- 2. per-(src,type) out-degree counts: one atomic per edge ----
__global__ void count_kernel(const int* __restrict__ ei, const int* __restrict__ et,
                             unsigned* __restrict__ cnt) {
  int e = blockIdx.x * 256 + threadIdx.x;
  if (e < EE) {
    int s = ei[e];        // src row
    int t = et[e];
    atomicAdd(&cnt[s * 4 + t], 1u);
  }
}

// ---- 3. fused x' = x + (sum_t cnt_t * emb_t)/max(total,1), store bf16 ----
__global__ void build_xb_kernel(const float* __restrict__ x, const float* __restrict__ emb,
                                const unsigned* __restrict__ cnt, unsigned short* __restrict__ xb) {
  int n = blockIdx.x, c = threadIdx.x;
  float c0 = (float)cnt[n * 4 + 0], c1 = (float)cnt[n * 4 + 1];
  float c2 = (float)cnt[n * 4 + 2], c3 = (float)cnt[n * 4 + 3];
  float tot = c0 + c1 + c2 + c3;
  float add = (c0 * emb[c] + c1 * emb[256 + c] + c2 * emb[512 + c] + c3 * emb[768 + c])
              / fmaxf(tot, 1.0f);
  xb[(size_t)n * 256 + c] = f2bf(x[(size_t)n * 256 + c] + add);
}

// ---- 4. transpose+convert W[k][n] -> Wt[n][k] bf16 (B-fragment becomes contiguous) ----
__global__ void transw_kernel(const float* __restrict__ wq, const float* __restrict__ wk,
                              const float* __restrict__ wv, const float* __restrict__ ws,
                              unsigned short* __restrict__ wt) {
  int idx = blockIdx.x * 256 + threadIdx.x;        // 4*65536 total
  int mat = idx >> 16, rem = idx & 65535;
  int k = rem >> 8, n = rem & 255;
  const float* w = (mat == 0) ? wq : (mat == 1) ? wk : (mat == 2) ? wv : ws;
  wt[(size_t)mat * 65536 + (size_t)n * 256 + k] = f2bf(w[(size_t)k * 256 + n]);
}

// ---- 5. GEMM: out = xb @ W + b via v_wmma_f32_16x16x32_bf16 ----
// grid (3125, 2, 4), block 256 (8 waves); wave -> one 16x16 output tile.
__global__ void gemm_kernel(const unsigned short* __restrict__ xb,
                            const unsigned short* __restrict__ wt,
                            const float* __restrict__ bq, const float* __restrict__ bk,
                            const float* __restrict__ bv, const float* __restrict__ bs,
                            float* __restrict__ oq, float* __restrict__ ok,
                            float* __restrict__ ov, float* __restrict__ os) {
  int wave = threadIdx.x >> 5;
  int lane = threadIdx.x & 31;
  int m  = lane & 15;          // A row / B col / D col
  int kh = lane >> 4;          // lane-half selector
  int rowTile = blockIdx.x;
  int colTile = blockIdx.y * 8 + wave;
  int mat = blockIdx.z;

  const float* bias; float* outp;
  switch (mat) {
    case 0:  bias = bq; outp = oq; break;
    case 1:  bias = bk; outp = ok; break;
    case 2:  bias = bv; outp = ov; break;
    default: bias = bs; outp = os; break;
  }

  const unsigned short* arow = xb + (size_t)(rowTile * 16 + m) * 256;
  const unsigned short* brow = wt + (size_t)mat * 65536 + (size_t)(colTile * 16 + m) * 256;

  v8f acc = {};
  #pragma unroll
  for (int kk = 0; kk < 8; ++kk) {
    // A 16x32 bf16: element e -> K = kk*32 + 16*kh + e (contiguous)
    int ka = kk * 32 + 16 * kh;
    u16x16 au = *(const u16x16*)(arow + ka);
    // B 32x16 bf16 (transposed storage): element e -> K = kk*32 + 16*kh + e (contiguous)
    u16x16 bu = *(const u16x16*)(brow + kk * 32 + 16 * kh);
    v16bf af = __builtin_bit_cast(v16bf, au);
    v16bf bf = __builtin_bit_cast(v16bf, bu);
    acc = __builtin_amdgcn_wmma_f32_16x16x32_bf16(false, af, false, bf,
                                                  (short)0, acc, false, false);
  }

  float bcol = bias[colTile * 16 + m];
  #pragma unroll
  for (int r = 0; r < 8; ++r) {                    // D: VGPR r -> row r + 8*kh, col m
    int orow = rowTile * 16 + r + 8 * kh;
    outp[(size_t)orow * 256 + colTile * 16 + m] = acc[r] + bcol;
  }
}

// ---- 6. per-edge attention logits + segment max (one wave per edge) ----
__global__ void alpha_kernel(const int* __restrict__ ei, const float* __restrict__ q,
                             const float* __restrict__ k, float* __restrict__ alpha,
                             float* __restrict__ amax) {
  int e = blockIdx.x * 8 + (threadIdx.x >> 5);
  if (e >= EE) return;
  int lane = threadIdx.x & 31;
  int src = ei[e], dst = ei[EE + e];
  int h = lane >> 3, sub = lane & 7;               // 8 lanes per head, 8 chans per lane
  const float* qp = q + (size_t)dst * 256 + h * 64 + sub * 8;
  const float* kp = k + (size_t)src * 256 + h * 64 + sub * 8;
  float s = 0.0f;
  #pragma unroll
  for (int j = 0; j < 8; ++j) s += qp[j] * kp[j];
  s += __shfl_xor(s, 1, 32);
  s += __shfl_xor(s, 2, 32);
  s += __shfl_xor(s, 4, 32);
  if (sub == 0) {
    float a = s * 0.125f;                          // / sqrt(64)
    alpha[(size_t)e * 4 + h] = a;
    atomicMaxF32(&amax[dst * 4 + h], a);
  }
}

// ---- 7. softmax-weighted scatter of v[src] onto dst ----
__global__ void scatter_kernel(const int* __restrict__ ei, const float* __restrict__ v,
                               const float* __restrict__ alpha, const float* __restrict__ amax,
                               float* __restrict__ denom, float* __restrict__ wv) {
  int e = blockIdx.x * 8 + (threadIdx.x >> 5);
  if (e >= EE) return;
  int lane = threadIdx.x & 31;
  int src = ei[e], dst = ei[EE + e];
  int h = lane >> 3, sub = lane & 7;
  float a = expf(alpha[(size_t)e * 4 + h] - amax[dst * 4 + h]);
  if (sub == 0) atomicAdd(&denom[dst * 4 + h], a);
  const float* vp = v  + (size_t)src * 256 + h * 64 + sub * 8;
  float*       wp = wv + (size_t)dst * 256 + h * 64 + sub * 8;
  #pragma unroll
  for (int j = 0; j < 8; ++j) atomicAdd(&wp[j], a * vp[j]);
}

// ---- 8. out = skip (already in d_out) + wv/denom ----
__global__ void finalize_kernel(const float* __restrict__ wv, const float* __restrict__ denom,
                                float* __restrict__ out) {
  int idx = blockIdx.x * 256 + threadIdx.x;
  if (idx >= NN * HC) return;
  int n = idx >> 8;
  int h = (idx >> 6) & 3;
  out[idx] += wv[idx] / (denom[n * 4 + h] + 1e-16f);
}

extern "C" void kernel_launch(void* const* d_in, const int* in_sizes, int n_in,
                              void* d_out, int out_size, void* d_ws, size_t ws_size,
                              hipStream_t stream) {
  const float* x   = (const float*)d_in[0];
  const int*   ei  = (const int*)  d_in[1];
  const int*   et  = (const int*)  d_in[2];
  const float* emb = (const float*)d_in[3];
  const float* Wq  = (const float*)d_in[4];
  const float* bq  = (const float*)d_in[5];
  const float* Wk  = (const float*)d_in[6];
  const float* bk  = (const float*)d_in[7];
  const float* Wv  = (const float*)d_in[8];
  const float* bv  = (const float*)d_in[9];
  const float* Ws  = (const float*)d_in[10];
  const float* bs  = (const float*)d_in[11];
  float* out = (float*)d_out;

  char* ws = (char*)d_ws;
  unsigned*       cnt   = (unsigned*)      (ws + OFF_CNT);
  unsigned short* xb    = (unsigned short*)(ws + OFF_XB);
  unsigned short* wt    = (unsigned short*)(ws + OFF_WT);
  float*          q     = (float*)         (ws + OFF_Q);
  float*          k     = (float*)         (ws + OFF_K);
  float*          v     = (float*)         (ws + OFF_V);
  float*          alpha = (float*)         (ws + OFF_ALPHA);
  float*          amax  = (float*)         (ws + OFF_AMAX);
  float*          den   = (float*)         (ws + OFF_DEN);
  float*          wv    = (float*)         (ws + OFF_WV);

  init_kernel<<<(NN * HC) / 256, 256, 0, stream>>>(cnt, den, amax, wv);
  count_kernel<<<(EE + 255) / 256, 256, 0, stream>>>(ei, et, cnt);
  build_xb_kernel<<<NN, 256, 0, stream>>>(x, emb, cnt, xb);
  transw_kernel<<<(4 * 256 * 256) / 256, 256, 0, stream>>>(Wq, Wk, Wv, Ws, wt);
  gemm_kernel<<<dim3(NN / 16, 2, 4), 256, 0, stream>>>(xb, wt, bq, bk, bv, bs,
                                                        q, k, v, out);
  alpha_kernel<<<EE / 8, 256, 0, stream>>>(ei, q, k, alpha, amax);
  scatter_kernel<<<EE / 8, 256, 0, stream>>>(ei, v, alpha, amax, den, wv);
  finalize_kernel<<<(NN * HC) / 256, 256, 0, stream>>>(wv, den, out);
}
